// GCN3layer_41901700939839
// MI455X (gfx1250) — matrix operand order repaired and checked by reading, not compiled, not run on hardware
//
#include <hip/hip_runtime.h>

typedef __attribute__((ext_vector_type(16))) __bf16 v16bf;
typedef __attribute__((ext_vector_type(8)))  __bf16 v8bf;
typedef __attribute__((ext_vector_type(8)))  float  v8f;

union V16 { v16bf v; struct { v8bf lo; v8bf hi; } h; };

// ---------------- degree / norm ----------------
__global__ void degree_init(float* deg, int N) {
    int i = blockIdx.x * blockDim.x + threadIdx.x;
    if (i < N) deg[i] = 1.0f;                      // self-loop contribution
}

__global__ void degree_accum(const int* __restrict__ dst, float* __restrict__ deg, int E) {
    int i = blockIdx.x * blockDim.x + threadIdx.x;
    if (i < E) atomicAdd(&deg[dst[i]], 1.0f);
}

__global__ void deg_to_dinv(float* deg, int N) {
    int i = blockIdx.x * blockDim.x + threadIdx.x;
    if (i < N) deg[i] = __frsqrt_rn(deg[i]);       // deg >= 1 always (self-loops)
}

// ---------------- precision conversion ----------------
__global__ void cvt_f32_to_bf16(const float* __restrict__ in, __bf16* __restrict__ out, long n) {
    long i = (long)blockIdx.x * blockDim.x + threadIdx.x;
    if (i < n) out[i] = (__bf16)in[i];
}

// W is [K][Ncol] row-major f32 -> Wt is [Ncol][K] row-major bf16
__global__ void transpose_to_bf16(const float* __restrict__ W, __bf16* __restrict__ Wt,
                                  int K, int Ncol) {
    int t = blockIdx.x * blockDim.x + threadIdx.x;
    if (t >= K * Ncol) return;
    int n = t / K, k = t - n * K;
    Wt[t] = (__bf16)W[k * Ncol + n];
}

// ---------------- WMMA GEMM: C[M,Ncol] = A[M,K](bf16) * Bt[Ncol,K](bf16)^T ----------------
// one wave -> one 16x16 tile; 8 waves per block
__global__ __launch_bounds__(256)
void gemm_bf16_wmma(const __bf16* __restrict__ A, const __bf16* __restrict__ Bt,
                    const float* __restrict__ bias, float* __restrict__ C,
                    int M, int K, int Ncol, int fuse_bias_relu) {
    const int lane  = threadIdx.x & 31;
    const int wave  = threadIdx.x >> 5;
    const int tileM = (blockIdx.x * 8 + wave) * 16;
    if (tileM + 16 > M) return;                    // wave-uniform: EXEC stays all-1 for WMMA
    const int tileN = blockIdx.y * 16;
    const int r15   = lane & 15;

    const long arow = (long)(tileM + r15) * K;     // A row for this lane (M = lane&15)
    const long brow = (long)(tileN + r15) * K;     // Bt row for this lane (N = lane&15)
    const int  kA   = (lane < 16) ? 0 : 16;        // A: K 0-15 vs 16-31 contiguous
    const int  kB   = (lane < 16) ? 0 : 16;        // B: K 0-15 vs 16-31 contiguous

    v8f acc = {};
#pragma unroll 4
    for (int kt = 0; kt < K; kt += 32) {
        V16 a, b;
        a.h.lo = *(const v8bf*)(A + arow + kt + kA);
        a.h.hi = *(const v8bf*)(A + arow + kt + kA + 8);
        b.h.lo = *(const v8bf*)(Bt + brow + kt + kB);
        b.h.hi = *(const v8bf*)(Bt + brow + kt + kB + 8);
        acc = __builtin_amdgcn_wmma_f32_16x16x32_bf16(
            false, a.v, false, b.v, (short)0, acc, false, false);
    }

    const int mOff = (lane < 16) ? 0 : 8;          // C/D: VGPR r -> M = mOff + r
    const int col  = tileN + r15;
    float bv = fuse_bias_relu ? bias[col] : 0.0f;
#pragma unroll
    for (int r = 0; r < 8; ++r) {
        float v = acc[r];
        if (fuse_bias_relu) v = fmaxf(v + bv, 0.0f);
        C[(long)(tileM + mOff + r) * Ncol + col] = v;
    }
}

// ---------------- edge scatter: agg[dst] += norm * h[src] ----------------
// one wave per edge; H=128 -> one float4 per lane, 4 f32 atomics (L2-resident)
__global__ __launch_bounds__(256)
void scatter_edges(const float* __restrict__ h, const int* __restrict__ src,
                   const int* __restrict__ dst, const float* __restrict__ dinv,
                   float* __restrict__ agg, int E, int H) {
    const int e = blockIdx.x * 8 + (threadIdx.x >> 5);
    if (e >= E) return;
    const int lane = threadIdx.x & 31;
    const int s = src[e], d = dst[e];
    const float nrm = dinv[s] * dinv[d];
    const float4* hs = (const float4*)(h + (long)s * H);
    float* ad = agg + (long)d * H;
    for (int i = lane; i < (H >> 2); i += 32) {
        float4 v = hs[i];
        atomicAdd(ad + 4 * i + 0, v.x * nrm);
        atomicAdd(ad + 4 * i + 1, v.y * nrm);
        atomicAdd(ad + 4 * i + 2, v.z * nrm);
        atomicAdd(ad + 4 * i + 3, v.w * nrm);
    }
}

// ---------------- epilogue: out = relu(agg + dinv^2 * h_lin + bias) -> bf16 ----------------
__global__ void post_bias_relu(const float* __restrict__ agg, const float* __restrict__ hlin,
                               const float* __restrict__ dinv, const float* __restrict__ bias,
                               __bf16* __restrict__ hb, int N, int H) {
    long t = (long)blockIdx.x * blockDim.x + threadIdx.x;
    if (t >= (long)N * H) return;
    int n = (int)(t / H);
    int k = (int)(t - (long)n * H);
    float di = dinv[n];
    float v = agg[t] + di * di * hlin[t] + bias[k];
    hb[t] = (__bf16)fmaxf(v, 0.0f);
}

extern "C" void kernel_launch(void* const* d_in, const int* in_sizes, int n_in,
                              void* d_out, int out_size, void* d_ws, size_t ws_size,
                              hipStream_t stream) {
    const float* x  = (const float*)d_in[0];
    const int*   ei = (const int*)d_in[1];
    const float* W1 = (const float*)d_in[2];
    const float* b1 = (const float*)d_in[3];
    const float* W2 = (const float*)d_in[4];
    const float* b2 = (const float*)d_in[5];
    const float* Wl = (const float*)d_in[6];
    const float* bl = (const float*)d_in[7];

    const int H = in_sizes[3];           // 128
    const int O = in_sizes[7];           // 64
    const int D = in_sizes[2] / H;       // 128
    const int N = in_sizes[0] / D;       // 100000
    const int E = in_sizes[1] / 2;       // 1.6M
    const int* srcIdx = ei;
    const int* dstIdx = ei + E;
    const int Dm = (D > H) ? D : H;

    // workspace carve (256B aligned slices)
    char* p = (char*)d_ws;
    auto carve = [&](size_t bytes) { void* q = (void*)p; p += (bytes + 255) & ~(size_t)255; return q; };
    float*  dinv = (float*)carve((size_t)N * 4);
    __bf16* xb   = (__bf16*)carve((size_t)N * Dm * 2);   // input acts (reused across layers)
    float*  hlin = (float*)carve((size_t)N * H * 4);     // GEMM output (pre-aggregation)
    float*  agg  = (float*)carve((size_t)N * H * 4);     // scatter accumulator
    __bf16* w1t  = (__bf16*)carve((size_t)D * H * 2);
    __bf16* w2t  = (__bf16*)carve((size_t)H * H * 2);
    __bf16* wlt  = (__bf16*)carve((size_t)H * O * 2);

    const int TB = 256;
    // 1) degree -> dinv
    degree_init<<<(N + TB - 1) / TB, TB, 0, stream>>>(dinv, N);
    degree_accum<<<(E + TB - 1) / TB, TB, 0, stream>>>(dstIdx, dinv, E);
    deg_to_dinv<<<(N + TB - 1) / TB, TB, 0, stream>>>(dinv, N);

    // 2) precision prep
    long nxd = (long)N * D;
    cvt_f32_to_bf16<<<(int)((nxd + TB - 1) / TB), TB, 0, stream>>>(x, xb, nxd);
    transpose_to_bf16<<<(D * H + TB - 1) / TB, TB, 0, stream>>>(W1, w1t, D, H);
    transpose_to_bf16<<<(H * H + TB - 1) / TB, TB, 0, stream>>>(W2, w2t, H, H);
    transpose_to_bf16<<<(H * O + TB - 1) / TB, TB, 0, stream>>>(Wl, wlt, H, O);

    const int tilesM   = N / 16;
    const dim3 gGemmH((tilesM + 7) / 8, H / 16);
    const dim3 gGemmO((tilesM + 7) / 8, O / 16);
    const int  eBlocks = (E + 7) / 8;
    const long nh      = (long)N * H;

    // ---- layer 1 ----
    gemm_bf16_wmma<<<gGemmH, TB, 0, stream>>>(xb, w1t, nullptr, hlin, N, D, H, 0);
    hipMemsetAsync(agg, 0, (size_t)N * H * 4, stream);
    scatter_edges<<<eBlocks, TB, 0, stream>>>(hlin, srcIdx, dstIdx, dinv, agg, E, H);
    post_bias_relu<<<(int)((nh + TB - 1) / TB), TB, 0, stream>>>(agg, hlin, dinv, b1, xb, N, H);

    // ---- layer 2 ----
    gemm_bf16_wmma<<<gGemmH, TB, 0, stream>>>(xb, w2t, nullptr, hlin, N, H, H, 0);
    hipMemsetAsync(agg, 0, (size_t)N * H * 4, stream);
    scatter_edges<<<eBlocks, TB, 0, stream>>>(hlin, srcIdx, dstIdx, dinv, agg, E, H);
    post_bias_relu<<<(int)((nh + TB - 1) / TB), TB, 0, stream>>>(agg, hlin, dinv, b2, xb, N, H);

    // ---- layer 3: plain linear + bias + relu, fused, straight to d_out ----
    gemm_bf16_wmma<<<gGemmO, TB, 0, stream>>>(xb, wlt, bl, (float*)d_out, N, H, O, 1);
}
